// MyVectorQuantizer_64398739636749
// MI455X (gfx1250) — matrix operand order repaired and checked by reading, not compiled, not run on hardware
//
#include <hip/hip_runtime.h>

typedef __attribute__((ext_vector_type(16))) __bf16 v16bf;
typedef __attribute__((ext_vector_type(8)))  float  v8f;
typedef unsigned short u16;
typedef unsigned int   u32;

// Problem sizes (from reference)
constexpr int  Bn = 8, Nn = 2048, Dn = 512, Kn = 8192;
constexpr int  Mtot = Bn * Nn;               // 16384 rows of z
constexpr float DECAY = 0.99f, ONE_MINUS = 0.01f, COMMITW = 0.25f, EPSV = 1e-5f;

// d_out layout (concatenated return tuple, all as float)
constexpr long long SZ_STE  = (long long)Mtot * Dn;          // 8,388,608
constexpr long long O_STE   = 0;
constexpr long long O_IDX   = O_STE + SZ_STE;                // 8,388,608
constexpr long long O_LOSS  = O_IDX + Mtot;                  // 8,404,992
constexpr long long O_SIZES = O_LOSS + 1;                    // 8,404,993
constexpr long long O_CB    = O_SIZES + Kn;                  // 8,413,185
constexpr long long O_EMB   = O_CB + (long long)Kn * Dn;     // 12,607,489 (16MB: used as bf16 scratch until embed_kernel)

// ---------- helpers: bf16 hi/lo split (round-to-nearest-even) ----------
__device__ __forceinline__ u16 f2bf(float f) {
    union { float f; u32 u; } c; c.f = f;
    u32 r = c.u + 0x7FFFu + ((c.u >> 16) & 1u);
    return (u16)(r >> 16);
}
__device__ __forceinline__ float bf2f(u16 h) {
    union { u32 u; float f; } c; c.u = ((u32)h) << 16;
    return c.f;
}

// ---------- kernel 1: e_sq[k] = sum_d e[k][d]^2 ----------
__global__ __launch_bounds__(256) void esq_kernel(const float* __restrict__ e,
                                                  float* __restrict__ esq) {
    int wave = (blockIdx.x * 256 + threadIdx.x) >> 5;
    int lane = threadIdx.x & 31;
    if (wave >= Kn) return;
    const float* row = e + (long long)wave * Dn;
    float s = 0.f;
    for (int d = lane; d < Dn; d += 32) { float v = row[d]; s += v * v; }
#pragma unroll
    for (int off = 16; off >= 1; off >>= 1) s += __shfl_xor(s, off, 32);
    if (lane == 0) esq[wave] = s;
}

// ---------- kernel 1b: pack E into bf16 hi/lo (one-time conversion) ----------
__global__ void pack_e_kernel(const float* __restrict__ e,
                              u16* __restrict__ eh, u16* __restrict__ el) {
    long long stride = (long long)gridDim.x * blockDim.x;
    for (long long i = (long long)blockIdx.x * blockDim.x + threadIdx.x;
         i < (long long)Kn * Dn; i += stride) {
        float f = e[i];
        u16 h = f2bf(f);
        eh[i] = h;
        el[i] = f2bf(f - bf2f(h));
    }
}

// ---------- kernel 2: init outputs (STE copy, loss=0, EMA decay seeds) ----------
__global__ void init_kernel(const float* __restrict__ z,
                            const float* __restrict__ sizes_in,
                            const float* __restrict__ cb_in,
                            float* __restrict__ out) {
    const long long total = SZ_STE + 1 + Kn + (long long)Kn * Dn;
    long long stride = (long long)gridDim.x * blockDim.x;
    for (long long i = (long long)blockIdx.x * blockDim.x + threadIdx.x;
         i < total; i += stride) {
        if (i < SZ_STE) {
            out[O_STE + i] = z[i];                      // STE output == input numerically
        } else if (i == SZ_STE) {
            out[O_LOSS] = 0.f;
        } else if (i < SZ_STE + 1 + Kn) {
            long long k = i - SZ_STE - 1;
            out[O_SIZES + k] = DECAY * sizes_in[k];
        } else {
            long long j = i - SZ_STE - 1 - Kn;
            out[O_CB + j] = DECAY * cb_in[j];
        }
    }
}

// ---------- kernel 3: fused WMMA GEMM + argmin ----------
constexpr int MT  = 32;        // rows per block (2 wave-rows of 16)
constexpr int NT  = 64;        // e-columns per chunk (4 wave-cols of 16)
constexpr int DC  = 128;       // D-slab per LDS stage
constexpr int ZLP = Dn + 8;    // 520: padded resident z row (1040B, 16B-aligned rows)
constexpr int ELP = DC + 8;    // 136: padded e slab row (272B, 16B-aligned rows)

// dynamic LDS layout (ushort units)
constexpr int ZS_N    = MT * ZLP;            // 16640 per array
constexpr int ES_N    = NT * ELP;            //  8704 per array
constexpr int OFF_ZH  = 0;
constexpr int OFF_ZL  = ZS_N;
constexpr int OFF_E0  = 2 * ZS_N;            // buffer0: hi then lo
constexpr int OFF_E1  = OFF_E0 + 2 * ES_N;
constexpr int OFF_RED = OFF_E1 + 2 * ES_N;   // byte offset 136192 (16B aligned)
constexpr size_t SMEM_BYTES = (size_t)OFF_RED * 2 + (size_t)MT * 4 * 4 * 2; // +red_v/red_i

union FragA { v16bf v; uint4 q[2]; };

// one async 16-byte global->LDS copy per lane (wave moves 512B, tracked by ASYNCcnt)
__device__ __forceinline__ void async_b128(u32 lds_off, u32 goff, const void* gbase) {
    asm volatile("global_load_async_to_lds_b128 %0, %1, %2"
                 :: "v"(lds_off), "v"(goff), "s"(gbase) : "memory");
}
#define WAIT_ASYNC(n) asm volatile("s_wait_asynccnt %0" :: "n"(n) : "memory")

// stage one 64x128 E slab (hi+lo) into LDS buffer via async DMA: 8 instrs/wave
__device__ __forceinline__ void issue_e_stage(const u16* __restrict__ gh,
                                              const u16* __restrict__ gl,
                                              u32 es_base, int colbase, int dg, int tid) {
#pragma unroll
    for (int i = 0; i < 4; ++i) {
        int t = tid + i * 256;              // 0..1023 chunk id
        int r = t >> 4, c = t & 15;         // row 0..63, 16B-chunk 0..15
        u32 lds_off = es_base + (u32)(r * ELP + c * 8) * 2;
        u32 goff    = ((u32)(colbase + r) * Dn + (u32)dg + (u32)c * 8) * 2;
        async_b128(lds_off, goff, gh);
        async_b128(lds_off + (u32)ES_N * 2, goff, gl);
    }
}

__global__ __launch_bounds__(256) void vq_argmin_kernel(
    const float* __restrict__ z,
    const u16* __restrict__ eh, const u16* __restrict__ el,
    const float* __restrict__ esq, int* __restrict__ nearest,
    float* __restrict__ out_idx)
{
    extern __shared__ __align__(16) u16 smem[];
    u16* zs_h = smem + OFF_ZH;
    u16* zs_l = smem + OFF_ZL;
    const u16* e_ptr[2] = { smem + OFF_E0, smem + OFF_E1 };
    u32 e_base[2] = { (u32)(uintptr_t)(smem + OFF_E0), (u32)(uintptr_t)(smem + OFF_E1) };
    float* red_v = (float*)(smem + OFF_RED);     // [MT][4]
    int*   red_i = (int*)(red_v + MT * 4);       // [MT][4]

    const int tid  = threadIdx.x;
    const int lane = tid & 31;
    const int wv   = tid >> 5;        // 0..7
    const int mw   = wv >> 2;         // 0..1  (wave row)
    const int nw   = wv & 3;          // 0..3  (wave col)
    const int mrow = mw * 16;
    const int m0   = blockIdx.x * MT;
    const int nloc = lane & 15;       // M (for A) / N (for B,C) within tile
    const int half = lane >> 4;       // lane half selects K sub-range

    // kick off DMA of E stage 0 while we stage Z
    issue_e_stage(eh, el, e_base[0], 0, 0, tid);

    // stage resident Z tile (32 x 512) as bf16 hi/lo -- once per block
    for (int idx = tid; idx < MT * Dn; idx += 256) {
        int r = idx >> 9, d = idx & (Dn - 1);
        float f = z[(long long)(m0 + r) * Dn + d];
        u16 h = f2bf(f);
        zs_h[r * ZLP + d] = h;
        zs_l[r * ZLP + d] = f2bf(f - bf2f(h));
    }
    __syncthreads();

    float bestv[8];
    int   besti[8];
#pragma unroll
    for (int j = 0; j < 8; ++j) { bestv[j] = 3.4e38f; besti[j] = 0; }

    for (int cc = 0; cc < Kn / NT; ++cc) {
        const int colbase = cc * NT;
        v8f acc = {};
        for (int dc = 0; dc < Dn / DC; ++dc) {
            const int s = cc * 4 + dc;
            // prefetch next slab into the other buffer, then wait for this slab's DMA
            if (s + 1 < (Kn / NT) * (Dn / DC)) {
                issue_e_stage(eh, el, e_base[(s + 1) & 1],
                              ((s + 1) >> 2) * NT, ((s + 1) & 3) * DC, tid);
                WAIT_ASYNC(8);      // my 8 copies for slab s are done (prefetch in flight)
            } else {
                WAIT_ASYNC(0);
            }
            __syncthreads();        // slab s visible from all waves
            const u16* esh = e_ptr[s & 1];
            const u16* esl = esh + ES_N;
            const int dgz = dc * DC;
#pragma unroll
            for (int kk = 0; kk < DC / 32; ++kk) {
                FragA ah, al, bh, bl;
                // A 16x32 bf16: lane half selects K quarters {0-7,16-23} / {8-15,24-31}
                const int za = (mrow + nloc) * ZLP + dgz + kk * 32 + half * 8;
                ah.q[0] = *(const uint4*)&zs_h[za];
                ah.q[1] = *(const uint4*)&zs_h[za + 16];
                al.q[0] = *(const uint4*)&zs_l[za];
                al.q[1] = *(const uint4*)&zs_l[za + 16];
                // B 32x16 bf16: lane half selects K quarters {0-7,16-23} / {8-15,24-31}; N=lane%16
                const int ea = (nw * 16 + nloc) * ELP + kk * 32 + half * 8;
                bh.q[0] = *(const uint4*)&esh[ea];
                bh.q[1] = *(const uint4*)&esh[ea + 16];
                bl.q[0] = *(const uint4*)&esl[ea];
                bl.q[1] = *(const uint4*)&esl[ea + 16];
                // z.e ~= zh.eh + zh.el + zl.eh  (~fp32 accuracy)
                acc = __builtin_amdgcn_wmma_f32_16x16x32_bf16(false, ah.v, false, bh.v,
                                                              (short)0, acc, false, false);
                acc = __builtin_amdgcn_wmma_f32_16x16x32_bf16(false, ah.v, false, bl.v,
                                                              (short)0, acc, false, false);
                acc = __builtin_amdgcn_wmma_f32_16x16x32_bf16(false, al.v, false, bh.v,
                                                              (short)0, acc, false, false);
            }
            __syncthreads();        // done reading buffer before its DMA reuse
        }
        // dist = e_sq[col] - 2*(z.e)  (drop constant z_sq for argmin)
        const int col = colbase + nw * 16 + nloc;
        const float eq = esq[col];
#pragma unroll
        for (int j = 0; j < 8; ++j) {
            float dist = eq - 2.0f * acc[j];
            if (dist < bestv[j]) { bestv[j] = dist; besti[j] = col; }
        }
    }

    // reduce argmin across 16 lanes (each lane holds one N column's best)
#pragma unroll
    for (int j = 0; j < 8; ++j) {
        float v = bestv[j]; int ix = besti[j];
#pragma unroll
        for (int off = 8; off >= 1; off >>= 1) {
            float ov = __shfl_xor(v, off, 16);
            int   oi = __shfl_xor(ix, off, 16);
            if (ov < v || (ov == v && oi < ix)) { v = ov; ix = oi; }
        }
        if (nloc == 0) {
            int row = mrow + j + half * 8;   // C layout: lanes16-31 hold M=j+8
            red_v[row * 4 + nw] = v;
            red_i[row * 4 + nw] = ix;
        }
    }
    __syncthreads();
    // combine the 4 wave-columns per row
    if (tid < MT) {
        float v = red_v[tid * 4]; int ix = red_i[tid * 4];
#pragma unroll
        for (int t = 1; t < 4; ++t) {
            float ov = red_v[tid * 4 + t]; int oi = red_i[tid * 4 + t];
            if (ov < v || (ov == v && oi < ix)) { v = ov; ix = oi; }
        }
        nearest[m0 + tid] = ix;
        out_idx[m0 + tid] = (float)ix;
    }
}

// ---------- kernel 4: EMA scatter + commitment loss ----------
__global__ __launch_bounds__(128) void scatter_kernel(
    const float* __restrict__ z, const float* __restrict__ e,
    const int* __restrict__ nearest, float* __restrict__ out)
{
    __shared__ float part[4];
    const int r = blockIdx.x;
    const int tid = threadIdx.x;
    const int k = nearest[r];
    const float* zr = z + (long long)r * Dn;
    const float* er = e + (long long)k * Dn;
    float* cb = out + O_CB + (long long)k * Dn;
    float local = 0.f;
    for (int d = tid; d < Dn; d += 128) {
        float zv = zr[d];
        float ev = er[d];           // quantized = e[nearest]
        float df = zv - ev;
        local += df * df;
        atomicAdd(&cb[d], ONE_MINUS * zv);   // EMA weighted_sum contribution
    }
#pragma unroll
    for (int off = 16; off >= 1; off >>= 1) local += __shfl_xor(local, off, 32);
    if ((tid & 31) == 0) part[tid >> 5] = local;
    __syncthreads();
    if (tid == 0) {
        float s = part[0] + part[1] + part[2] + part[3];
        atomicAdd(out + O_LOSS, s * (COMMITW / (float)SZ_STE));
        atomicAdd(out + O_SIZES + k, ONE_MINUS);
    }
}

// ---------- kernel 5: new_embedding = codebook / (sizes + eps) ----------
// (also overwrites the O_EMB region that served as bf16 E scratch)
__global__ void embed_kernel(float* __restrict__ out) {
    long long i = (long long)blockIdx.x * blockDim.x + threadIdx.x;
    if (i >= (long long)Kn * Dn) return;
    int k = (int)(i >> 9);   // / Dn
    out[O_EMB + i] = out[O_CB + i] / (out[O_SIZES + k] + EPSV);
}

extern "C" void kernel_launch(void* const* d_in, const int* in_sizes, int n_in,
                              void* d_out, int out_size, void* d_ws, size_t ws_size,
                              hipStream_t stream) {
    const float* z        = (const float*)d_in[0];  // (B,N,D)
    const float* e        = (const float*)d_in[1];  // (K,D)
    const float* sizes_in = (const float*)d_in[2];  // (K,)
    const float* cb_in    = (const float*)d_in[3];  // (K,D)
    float* out = (float*)d_out;

    float* esq     = (float*)d_ws;                              // K floats
    int*   nearest = (int*)((char*)d_ws + Kn * sizeof(float));  // Mtot ints

    // bf16 hi/lo packed E lives in the O_EMB output region (dead until embed_kernel)
    u16* eh = (u16*)(out + O_EMB);
    u16* el = eh + (long long)Kn * Dn;

    esq_kernel<<<Kn / 8, 256, 0, stream>>>(e, esq);
    pack_e_kernel<<<2048, 256, 0, stream>>>(e, eh, el);
    init_kernel<<<4096, 256, 0, stream>>>(z, sizes_in, cb_in, out);
    vq_argmin_kernel<<<Mtot / MT, 256, SMEM_BYTES, stream>>>(z, eh, el, esq, nearest,
                                                             out + O_IDX);
    scatter_kernel<<<Mtot, 128, 0, stream>>>(z, e, nearest, out);
    embed_kernel<<<(Kn * Dn + 255) / 256, 256, 0, stream>>>(out);
}
